// MPNNLayer_80290118631446
// MI455X (gfx1250) — compile-verified, matrix operand order checked by reading, not run on hardware
//
#include <hip/hip_runtime.h>

typedef __attribute__((ext_vector_type(16))) _Float16 v16h;
typedef __attribute__((ext_vector_type(8)))  float    v8f;

#define HF  128
#define EDF 32
#define K1  288
#define N1  256

union Frag16 { v16h h; unsigned int u[8]; };

__device__ __forceinline__ float gelu_exact(float v) {
    return 0.5f * v * (1.0f + erff(v * 0.70710678118654752440f));
}

// ---------------------------------------------------------------------------
// Cooperative global->LDS slab copy, 16B per lane per op.
// Uses CDNA5 async-to-LDS (ASYNCcnt) when the builtin exists, else a plain
// synchronous copy through VGPRs.
// Builtin signature (from hipcc diagnostic): param0 = global (AS1) int4*,
// param1 = LDS (AS3) int4*, then imm offset, imm cpol.
// ---------------------------------------------------------------------------
#if __has_builtin(__builtin_amdgcn_global_load_async_to_lds_b128)
#define ASYNC_LDS_COPY 1
typedef int v4i_gcc __attribute__((vector_size(16)));
typedef __attribute__((address_space(1))) v4i_gcc* gptr16_t;
typedef __attribute__((address_space(3))) v4i_gcc* lptr16_t;
#else
#define ASYNC_LDS_COPY 0
#endif

__device__ __forceinline__ void stage_slab(const _Float16* __restrict__ gsrc,
                                           _Float16* lds_dst, int bytes,
                                           int tid, int nthreads)
{
    const int chunks = bytes >> 4;          // 16B chunks
#if ASYNC_LDS_COPY
    for (int c = tid; c < chunks; c += nthreads)
        __builtin_amdgcn_global_load_async_to_lds_b128(
            (gptr16_t)((const char*)gsrc + (size_t)c * 16),
            (lptr16_t)((char*)lds_dst + (size_t)c * 16), 0, 0);
#else
    for (int c = tid; c < chunks; c += nthreads)
        ((uint4*)lds_dst)[c] = ((const uint4*)gsrc)[c];
#endif
}

__device__ __forceinline__ void wait_async_done() {
#if ASYNC_LDS_COPY && __has_builtin(__builtin_amdgcn_s_wait_asynccnt)
    __builtin_amdgcn_s_wait_asynccnt(0);
#endif
}

// ---------------------------------------------------------------------------
// Pack f32 weight [Kdim x Ndim] into f16 WMMA B-fragment order: per (kt, nt)
// tile a 512-half block, lane L holds 16 halves in v_wmma_f32_16x16x32_f16
// B layout (lanes 0-15: N=L, K pairs 0..7 & 16..23; lanes 16-31: N=L-16,
// K pairs 8..15 & 24..31).
// ---------------------------------------------------------------------------
__global__ void pack_weight_kernel(const float* __restrict__ W,
                                   _Float16* __restrict__ dst,
                                   int Kdim, int Ndim)
{
    const int ntc = Ndim >> 4;
    const int total = (Kdim >> 5) * ntc * 512;
    int t = blockIdx.x * 256 + threadIdx.x;
    if (t >= total) return;
    const int blk = t >> 9, r = t & 511;
    const int lane = r >> 4, j = r & 15;
    const int nt = blk % ntc, kt = blk / ntc;
    const int laneHi = lane >> 4, lanelo = lane & 15;
    const int g = j >> 1, hb = j & 1;
    const int k = kt * 32 + laneHi * 8 +
                  ((g < 4) ? (2 * g + hb) : (16 + 2 * (g - 4) + hb));
    const int n = nt * 16 + lanelo;
    dst[t] = (_Float16)W[k * Ndim + n];
}

__global__ void zero_kernel(float* __restrict__ p, long long n) {
    long long i = (long long)blockIdx.x * 256 + threadIdx.x;
    if (i < n) p[i] = 0.0f;
}

// ---------------------------------------------------------------------------
// Edge message kernel: 64 edges per block, 8 waves.
//   m_in = [x[row] | x[col] | edge_attr]  (64 x 288, f16 in LDS)
//   h    = GELU(m_in @ Wm1 + bm1)         (64 x 256)   WMMA, async B slabs
//   msg  = h @ Wm2 + bm2                  (64 x 128)   WMMA, async B slabs
//   aggr[col] += msg ; cnt[col] += 1      (f32 global atomics)
// ---------------------------------------------------------------------------
__global__ void __launch_bounds__(256) edge_msg_kernel(
    const float* __restrict__ x, const long long* __restrict__ ei,
    const float* __restrict__ ea,
    const _Float16* __restrict__ wm1p, const float* __restrict__ bm1,
    const _Float16* __restrict__ wm2p, const float* __restrict__ bm2,
    float* __restrict__ aggr, float* __restrict__ cnt, int E)
{
    __shared__ __align__(16) _Float16 smem[64 * 296];   // 37,888 B, aliased
    __shared__ __align__(16) _Float16 sB[2 * 8192];     // 32 KB dbl-buffered B
    __shared__ long long sRow[64], sCol[64];            // cached edge endpoints
    _Float16* sA = smem;   // stage 1: 64 x 288, stride 296
    _Float16* sH = smem;   // stage 2: 64 x 256, stride 264

    const int tid = threadIdx.x;
    const int wave = tid >> 5, lane = tid & 31;
    const int laneHi = lane >> 4, lanelo = lane & 15;
    const long long e0 = (long long)blockIdx.x * 64;

    // ---- cache edge endpoint indices once ----
    if (tid < 64) {
        const long long e = e0 + tid;
        const long long ec = (e < E) ? e : (long long)(E - 1);
        sRow[tid] = ei[ec];
        sCol[tid] = ei[(size_t)E + ec];
        if (e < E) atomicAdd(&cnt[sCol[tid]], 1.0f);    // in-degree counters
    }
    // prologue: stage GEMM1 weight slab kt=0 (16 KB) asynchronously
    stage_slab(wm1p, sB, 16384, tid, 256);
    __syncthreads();

    // ---- gather + f16 convert into LDS ----
    for (int idx = tid; idx < 64 * K1; idx += 256) {
        const int em = idx / K1, c = idx - em * K1;
        const long long e = e0 + em;
        float v;
        if (c < HF)          v = x[sRow[em] * HF + c];
        else if (c < 2 * HF) v = x[sCol[em] * HF + (c - HF)];
        else {
            const long long ec = (e < E) ? e : (long long)(E - 1);
            v = ea[ec * EDF + (c - 2 * HF)];
        }
        sA[em * 296 + c] = (_Float16)((e < E) ? v : 0.0f);
    }

    // ---- GEMM1: 64x288 @ 288x256 ; wave -> mtile (wave&3), 8 ntiles ----
    const int mt = wave & 3;
    const int nb1 = (wave >> 2) * 8;
    v8f acc[8] = {};
    const int arow = (mt * 16 + lanelo) * 296;
    for (int kt = 0; kt < 9; ++kt) {
        wait_async_done();           // own async ops for slab kt complete
        __syncthreads();             // whole block: slab kt visible, kt-1 reads done
        const _Float16* bcur = sB + (kt & 1) * 8192;
        if (kt + 1 < 9)              // overlap: stage slab kt+1 while computing
            stage_slab(wm1p + (size_t)(kt + 1) * 8192,
                       sB + ((kt + 1) & 1) * 8192, 16384, tid, 256);
        Frag16 a;
        const int kbase = kt * 32 + laneHi * 8;
        #pragma unroll
        for (int g = 0; g < 8; ++g) {
            const int k = kbase + ((g < 4) ? (2 * g) : (16 + 2 * (g - 4)));
            a.u[g] = *(const unsigned int*)(sA + arow + k);
        }
        #pragma unroll
        for (int nt = 0; nt < 8; ++nt) {
            const v16h b = *(const v16h*)(bcur + (nb1 + nt) * 512 + lane * 16);
            acc[nt] = __builtin_amdgcn_wmma_f32_16x16x32_f16(
                false, a.h, false, b, (short)0, acc[nt], false, false);
        }
    }
    __syncthreads();                 // all sA reads done before aliasing

    // ---- epilogue: bias + exact GELU -> sH (f16) ----
    #pragma unroll
    for (int nt = 0; nt < 8; ++nt) {
        const int n = (nb1 + nt) * 16 + lanelo;
        const float bias = bm1[n];
        #pragma unroll
        for (int r = 0; r < 8; ++r) {
            const int m = mt * 16 + r + laneHi * 8;
            sH[m * 264 + n] = (_Float16)gelu_exact(acc[nt][r] + bias);
        }
    }
    // prologue for GEMM2 slabs (8 KB each); sB fully free after barrier above
    __syncthreads();
    stage_slab(wm2p, sB, 8192, tid, 256);

    // ---- GEMM2: 64x256 @ 256x128 ; wave -> mtile, 4 ntiles ----
    const int nb2 = (wave >> 2) * 4;
    v8f acc2[4] = {};
    const int arow2 = (mt * 16 + lanelo) * 264;
    for (int kt = 0; kt < 8; ++kt) {
        wait_async_done();
        __syncthreads();
        const _Float16* bcur = sB + (kt & 1) * 8192;
        if (kt + 1 < 8)
            stage_slab(wm2p + (size_t)(kt + 1) * 4096,
                       sB + ((kt + 1) & 1) * 8192, 8192, tid, 256);
        Frag16 a;
        const int kbase = kt * 32 + laneHi * 8;
        #pragma unroll
        for (int g = 0; g < 8; ++g) {
            const int k = kbase + ((g < 4) ? (2 * g) : (16 + 2 * (g - 4)));
            a.u[g] = *(const unsigned int*)(sH + arow2 + k);
        }
        #pragma unroll
        for (int nt = 0; nt < 4; ++nt) {
            const v16h b = *(const v16h*)(bcur + (nb2 + nt) * 512 + lane * 16);
            acc2[nt] = __builtin_amdgcn_wmma_f32_16x16x32_f16(
                false, a.h, false, b, (short)0, acc2[nt], false, false);
        }
    }

    // ---- scatter-add messages to destination nodes ----
    #pragma unroll
    for (int nt = 0; nt < 4; ++nt) {
        const int n = (nb2 + nt) * 16 + lanelo;
        const float bias = bm2[n];
        #pragma unroll
        for (int r = 0; r < 8; ++r) {
            const int m = mt * 16 + r + laneHi * 8;
            if (e0 + m < E)
                atomicAdd(&aggr[sCol[m] * HF + n], acc2[nt][r] + bias);
        }
    }
}

// ---------------------------------------------------------------------------
// Node update kernel: 64 nodes per block, 8 waves.
//   u_in = [x | aggr/max(cnt,1)]  (64 x 256 f16 in LDS)
//   u    = GELU(u_in @ Wu1 + bu1) @ Wu2 + bu2       (WMMA x2)
//   y    = LayerNorm(x + u) * gamma + beta
// ---------------------------------------------------------------------------
__global__ void __launch_bounds__(256) node_update_kernel(
    const float* __restrict__ x, const float* __restrict__ aggr,
    const float* __restrict__ cnt,
    const _Float16* __restrict__ wu1p, const float* __restrict__ bu1,
    const _Float16* __restrict__ wu2p, const float* __restrict__ bu2,
    const float* __restrict__ gamma, const float* __restrict__ beta,
    float* __restrict__ out, int N)
{
    __shared__ __align__(16) _Float16 smem[64 * 264];   // 33,792 B, aliased
    _Float16* sU = smem;           // 64 x 256 f16, stride 264
    _Float16* sG = smem;           // 64 x 128 f16, stride 136
    float*    sY = (float*)smem;   // 64 x 128 f32

    const int tid = threadIdx.x;
    const int wave = tid >> 5, lane = tid & 31;
    const int laneHi = lane >> 4, lanelo = lane & 15;
    const long long n0 = (long long)blockIdx.x * 64;

    if (tid < 16) __builtin_prefetch(wu1p + (size_t)tid * 2048, 0, 1);

    // ---- stage u_in = [x | mean-aggr] ----
    for (int idx = tid; idx < 64 * 256; idx += 256) {
        const int m = idx >> 8, c = idx & 255;
        const long long node = n0 + m;
        const long long nc = (node < N) ? node : (long long)(N - 1);
        float v;
        if (c < HF) v = x[nc * HF + c];
        else        v = aggr[nc * HF + (c - HF)] / fmaxf(cnt[nc], 1.0f);
        sU[m * 264 + c] = (_Float16)v;
    }
    __syncthreads();

    // ---- GEMM1: 64x256 @ 256x128 ----
    const int mt = wave & 3;
    const int nb = (wave >> 2) * 4;
    v8f acc[4] = {};
    const int arow = (mt * 16 + lanelo) * 264;
    for (int kt = 0; kt < 8; ++kt) {
        Frag16 a;
        const int kbase = kt * 32 + laneHi * 8;
        #pragma unroll
        for (int g = 0; g < 8; ++g) {
            const int k = kbase + ((g < 4) ? (2 * g) : (16 + 2 * (g - 4)));
            a.u[g] = *(const unsigned int*)(sU + arow + k);
        }
        #pragma unroll
        for (int nt = 0; nt < 4; ++nt) {
            const v16h b = *(const v16h*)(wu1p +
                ((size_t)(kt * 8 + nb + nt) << 9) + lane * 16);
            acc[nt] = __builtin_amdgcn_wmma_f32_16x16x32_f16(
                false, a.h, false, b, (short)0, acc[nt], false, false);
        }
    }
    __syncthreads();
    #pragma unroll
    for (int nt = 0; nt < 4; ++nt) {
        const int n = (nb + nt) * 16 + lanelo;
        const float bias = bu1[n];
        #pragma unroll
        for (int r = 0; r < 8; ++r) {
            const int m = mt * 16 + r + laneHi * 8;
            sG[m * 136 + n] = (_Float16)gelu_exact(acc[nt][r] + bias);
        }
    }
    __syncthreads();

    // ---- GEMM2: 64x128 @ 128x128 ----
    v8f acc2[4] = {};
    const int arow2 = (mt * 16 + lanelo) * 136;
    for (int kt = 0; kt < 4; ++kt) {
        Frag16 a;
        const int kbase = kt * 32 + laneHi * 8;
        #pragma unroll
        for (int g = 0; g < 8; ++g) {
            const int k = kbase + ((g < 4) ? (2 * g) : (16 + 2 * (g - 4)));
            a.u[g] = *(const unsigned int*)(sG + arow2 + k);
        }
        #pragma unroll
        for (int nt = 0; nt < 4; ++nt) {
            const v16h b = *(const v16h*)(wu2p +
                ((size_t)(kt * 8 + nb + nt) << 9) + lane * 16);
            acc2[nt] = __builtin_amdgcn_wmma_f32_16x16x32_f16(
                false, a.h, false, b, (short)0, acc2[nt], false, false);
        }
    }
    __syncthreads();               // sG reads done before aliasing as sY

    // ---- y = x + u -> sY (f32) ----
    #pragma unroll
    for (int nt = 0; nt < 4; ++nt) {
        const int n = (nb + nt) * 16 + lanelo;
        const float bias = bu2[n];
        #pragma unroll
        for (int r = 0; r < 8; ++r) {
            const int m = mt * 16 + r + laneHi * 8;
            const long long node = n0 + m;
            const long long nc = (node < N) ? node : (long long)(N - 1);
            sY[m * 128 + n] = acc2[nt][r] + bias + x[nc * HF + n];
        }
    }
    __syncthreads();

    // ---- LayerNorm: each wave32 normalizes 8 rows ----
    #pragma unroll
    for (int i = 0; i < 8; ++i) {
        const int m = wave * 8 + i;
        float vals[4], s = 0.f, s2 = 0.f;
        #pragma unroll
        for (int q = 0; q < 4; ++q) {
            const float v = sY[m * 128 + lane + 32 * q];
            vals[q] = v; s += v; s2 += v * v;
        }
        #pragma unroll
        for (int off = 16; off > 0; off >>= 1) {
            s  += __shfl_xor(s,  off, 32);
            s2 += __shfl_xor(s2, off, 32);
        }
        const float mu  = s * (1.0f / 128.0f);
        const float inv = rsqrtf(s2 * (1.0f / 128.0f) - mu * mu + 1e-5f);
        const long long node = n0 + m;
        if (node < N) {
            #pragma unroll
            for (int q = 0; q < 4; ++q) {
                const int n = lane + 32 * q;
                out[node * HF + n] = (vals[q] - mu) * inv * gamma[n] + beta[n];
            }
        }
    }
}

// ---------------------------------------------------------------------------
extern "C" void kernel_launch(void* const* d_in, const int* in_sizes, int n_in,
                              void* d_out, int out_size, void* d_ws, size_t ws_size,
                              hipStream_t stream)
{
    const float*     x   = (const float*)d_in[0];
    const long long* ei  = (const long long*)d_in[1];
    const float*     ea  = (const float*)d_in[2];
    const float* wm1 = (const float*)d_in[3];
    const float* bm1 = (const float*)d_in[4];
    const float* wm2 = (const float*)d_in[5];
    const float* bm2 = (const float*)d_in[6];
    const float* wu1 = (const float*)d_in[7];
    const float* bu1 = (const float*)d_in[8];
    const float* wu2 = (const float*)d_in[9];
    const float* bu2 = (const float*)d_in[10];
    const float* gamma = (const float*)d_in[11];
    const float* beta  = (const float*)d_in[12];

    const int N = in_sizes[0] / HF;       // 50000
    const int E = in_sizes[2] / EDF;      // 800000

    char* ws = (char*)d_ws;
    size_t off = 0;
    _Float16* wm1p = (_Float16*)(ws + off); off += (size_t)9 * 16 * 512 * 2;  // 147,456
    _Float16* wm2p = (_Float16*)(ws + off); off += (size_t)8 * 8  * 512 * 2;  //  65,536
    _Float16* wu1p = (_Float16*)(ws + off); off += (size_t)8 * 8  * 512 * 2;  //  65,536
    _Float16* wu2p = (_Float16*)(ws + off); off += (size_t)4 * 8  * 512 * 2;  //  32,768
    float* aggr = (float*)(ws + off); off += (size_t)N * HF * 4;              // 25.6 MB
    float* cnt  = (float*)(ws + off); off += (size_t)N * 4;                   // contiguous after aggr

    // pack weights f32 -> f16 fragment order (tiny, once per call)
    pack_weight_kernel<<<(9 * 16 * 512) / 256, 256, 0, stream>>>(wm1, wm1p, K1, N1);
    pack_weight_kernel<<<(8 * 8  * 512) / 256, 256, 0, stream>>>(wm2, wm2p, 256, 128);
    pack_weight_kernel<<<(8 * 8  * 512) / 256, 256, 0, stream>>>(wu1, wu1p, 256, 128);
    pack_weight_kernel<<<(4 * 8  * 512) / 256, 256, 0, stream>>>(wu2, wu2p, 128, 128);

    // zero aggr + cnt (contiguous region) every call -> deterministic
    const long long zn = (long long)N * (HF + 1);
    zero_kernel<<<(int)((zn + 255) / 256), 256, 0, stream>>>(aggr, zn);

    edge_msg_kernel<<<(E + 63) / 64, 256, 0, stream>>>(
        x, ei, ea, wm1p, bm1, wm2p, bm2, aggr, cnt, E);
    node_update_kernel<<<(N + 63) / 64, 256, 0, stream>>>(
        x, aggr, cnt, wu1p, bu1, wu2p, bu2, gamma, beta, (float*)d_out, N);
}
